// TransformerBlock_51281909514645
// MI455X (gfx1250) — matrix operand order repaired and checked by reading, not compiled, not run on hardware
//
#include <hip/hip_runtime.h>
#include <hip/hip_bf16.h>
#include <math.h>

typedef __bf16 bf16;
typedef __attribute__((ext_vector_type(16))) __bf16 v16bf;
typedef __attribute__((ext_vector_type(8)))  __bf16 v8bf;
typedef __attribute__((ext_vector_type(8)))  float  v8f;

#define B_  4
#define S_  1024
#define D_  1024
#define H_  16
#define HD_ 64
#define E_  8
#define NT_ (B_ * S_)      // 4096 tokens / rows

__device__ __forceinline__ float gelu_new(float x) {
  const float c = 0.7978845608028654f;  // sqrt(2/pi)
  float x3 = x * x * x;
  return 0.5f * x * (1.0f + tanhf(c * (x + 0.044715f * x3)));
}

// ---------------- elementwise conversion ----------------
__global__ void cvt_f32_bf16(const float* __restrict__ src,
                             bf16* __restrict__ dst, int n) {
  int i = blockIdx.x * blockDim.x + threadIdx.x;
  int stride = gridDim.x * blockDim.x;
  for (; i < n; i += stride) dst[i] = (bf16)src[i];
}

// Pack Wq|Wk|Wv -> bf16 [D][3D], and bq|bk|bv -> f32 [3D]
__global__ void pack_qkv(const float* __restrict__ Wq, const float* __restrict__ Wk,
                         const float* __restrict__ Wv,
                         const float* __restrict__ bq, const float* __restrict__ bk,
                         const float* __restrict__ bv,
                         bf16* __restrict__ Wdst, float* __restrict__ bdst) {
  int i = blockIdx.x * blockDim.x + threadIdx.x;
  int stride = gridDim.x * blockDim.x;
  for (int idx = i; idx < D_ * 3 * D_; idx += stride) {
    int k = idx / (3 * D_);
    int j = idx % (3 * D_);
    float v = (j < D_)     ? Wq[k * D_ + j]
            : (j < 2 * D_) ? Wk[k * D_ + (j - D_)]
                           : Wv[k * D_ + (j - 2 * D_)];
    Wdst[idx] = (bf16)v;
  }
  if (i < 3 * D_) {
    bdst[i] = (i < D_) ? bq[i] : (i < 2 * D_) ? bk[i - D_] : bv[i - 2 * D_];
  }
}

// ---------------- bf16 WMMA GEMM (double-buffered) ----------------
// C[M,N] = A[M,K] @ B[K,N] (+bias[n]), epilogue by MODE:
//   MODE 0: Cf = acc + bias                       (fp32 out)
//   MODE 1: Ch = bf16(gelu_new(acc + bias))       (bf16 out)
//   MODE 2: Cf = gate * (acc + bias) [+= prior]   (fp32 out, gate-scaled)
// Block tile 128x128, K-step 32, 256 threads = 8 waves (4 x 2),
// each wave owns a 32x64 sub-tile = 2x4 WMMA 16x16 accumulators.
// Ping-pong LDS: global prefetch of tile k+1 overlaps WMMA on tile k,
// one barrier per k-step.
template <int MODE>
__global__ __launch_bounds__(256)
void gemm_bf16(const bf16* __restrict__ A, const bf16* __restrict__ Bm,
               const float* __restrict__ bias,
               float* __restrict__ Cf, bf16* __restrict__ Ch,
               const float* __restrict__ gates, int expert, int accumulate,
               int M, int N, int K) {
  constexpr int BM = 128, BN = 128, BK = 32;
  constexpr int AS = 40, BSS = 40;        // padded LDS strides (elements)
  __shared__ bf16 Alds[2][BM * AS];       // [m][k]
  __shared__ bf16 Blds[2][BN * BSS];      // [n][k] (transposed on fill)

  const int tid  = threadIdx.x;
  const int lane = tid & 31;
  const int w    = tid >> 5;
  const int wm   = w & 3;        // 0..3 -> m offset wm*32
  const int wn   = w >> 2;       // 0..1 -> n offset wn*64
  const int l15  = lane & 15;
  const int lhf  = lane >> 4;    // half-wave select
  const int m0   = blockIdx.y * BM;
  const int n0   = blockIdx.x * BN;

  // per-thread staging assignments
  const int arow0 = tid >> 2, akc = (tid & 3) * 8;   // A chunk 0
  const int arow1 = arow0 + 64;                      // A chunk 1
  const int bkk = (tid >> 4) * 2, bnc = (tid & 15) * 8;  // B k-pair chunk

  v8f acc[2][4];
#pragma unroll
  for (int i = 0; i < 2; i++)
#pragma unroll
    for (int j = 0; j < 4; j++)
#pragma unroll
      for (int q = 0; q < 8; q++) acc[i][j][q] = 0.0f;

  v8bf ra0, ra1, rb0, rb1;  // global prefetch registers

  auto gload = [&](int kt) {
    ra0 = *(const v8bf*)(A + (size_t)(m0 + arow0) * K + kt + akc);
    ra1 = *(const v8bf*)(A + (size_t)(m0 + arow1) * K + kt + akc);
    rb0 = *(const v8bf*)(Bm + (size_t)(kt + bkk) * N + n0 + bnc);
    rb1 = *(const v8bf*)(Bm + (size_t)(kt + bkk + 1) * N + n0 + bnc);
  };
  auto lstore = [&](int buf) {
    *(v8bf*)&Alds[buf][arow0 * AS + akc] = ra0;
    *(v8bf*)&Alds[buf][arow1 * AS + akc] = ra1;
#pragma unroll
    for (int j = 0; j < 8; j++) {  // pack k-pair into one dword store
      union { bf16 h[2]; unsigned u; } p;
      p.h[0] = rb0[j];
      p.h[1] = rb1[j];
      *(unsigned*)&Blds[buf][(bnc + j) * BSS + bkk] = p.u;
    }
  };

  const int nk = K / BK;
  gload(0);
  lstore(0);
  __syncthreads();

  for (int kt = 0; kt < nk; kt++) {
    if (kt + 1 < nk) gload((kt + 1) * BK);  // issue early, hide latency
    const int cur = kt & 1;

    // A fragment: lanes 0-15 rows m, K = {0..7,16..23}; lanes 16-31 K = {8..15,24..31}
    union { v16bf v; v8bf h[2]; } afr[2], bfr[4];
#pragma unroll
    for (int mt = 0; mt < 2; mt++) {
      int row = wm * 32 + mt * 16 + l15;
      int kb = lhf ? 8 : 0;
      afr[mt].h[0] = *(const v8bf*)&Alds[cur][row * AS + kb];
      afr[mt].h[1] = *(const v8bf*)&Alds[cur][row * AS + kb + 16];
    }
    // B fragment: lanes 0-15 cols n, K = {0..7,16..23}; lanes 16-31 K = {8..15,24..31}
#pragma unroll
    for (int nt = 0; nt < 4; nt++) {
      int col = wn * 64 + nt * 16 + l15;
      int kb = lhf ? 8 : 0;
      bfr[nt].h[0] = *(const v8bf*)&Blds[cur][col * BSS + kb];
      bfr[nt].h[1] = *(const v8bf*)&Blds[cur][col * BSS + kb + 16];
    }
#pragma unroll
    for (int mt = 0; mt < 2; mt++)
#pragma unroll
      for (int nt = 0; nt < 4; nt++)
        acc[mt][nt] = __builtin_amdgcn_wmma_f32_16x16x32_bf16(
            false, afr[mt].v, false, bfr[nt].v, (short)0, acc[mt][nt],
            false, false);

    if (kt + 1 < nk) {
      lstore((kt + 1) & 1);  // write other buffer; no barrier needed before
      __syncthreads();       // publish buffer (kt+1)&1, retire reads of cur
    }
  }

  // epilogue: acc element r of lane = C[row r + 8*lhf][col l15] of 16x16 tile
  const float g = (MODE == 2) ? gates[(m0 / S_) * E_ + expert] : 0.0f;
#pragma unroll
  for (int mt = 0; mt < 2; mt++) {
#pragma unroll
    for (int nt = 0; nt < 4; nt++) {
      int n = n0 + wn * 64 + nt * 16 + l15;
      float bn = bias[n];
#pragma unroll
      for (int r = 0; r < 8; r++) {
        int m = m0 + wm * 32 + mt * 16 + r + lhf * 8;
        float v = acc[mt][nt][r] + bn;
        size_t off = (size_t)m * N + n;
        if (MODE == 0) {
          Cf[off] = v;
        } else if (MODE == 1) {
          Ch[off] = (bf16)gelu_new(v);
        } else {
          float o = g * v;
          if (accumulate) o += Cf[off];
          Cf[off] = o;
        }
      }
    }
  }
}

// ---------------- per-token cross-head attention ----------------
// QKV: [B*S][3*D] fp32 rows (Q|K|V). Writes a in permuted layout as bf16:
// a[b][h*64 + s/16][(s%16)*64 + hd]
__global__ __launch_bounds__(256)
void attn_kernel(const float* __restrict__ QKV, bf16* __restrict__ A16) {
  __shared__ float q[D_], k[D_], v[D_];
  __shared__ float sc[H_][H_ + 1];
  __shared__ float at[H_][H_ + 1];
  int tok = blockIdx.x;
  int b = tok >> 10, s = tok & (S_ - 1);
  const float* base = QKV + (size_t)tok * (3 * D_);
  int t = threadIdx.x;
  for (int i = t; i < D_; i += 256) {
    q[i] = base[i];
    k[i] = base[D_ + i];
    v[i] = base[2 * D_ + i];
  }
  __syncthreads();
  {  // 16x16 score matrix across heads: one (qh,kh) per thread
    int qh = t >> 4, kh = t & 15;
    float acc = 0.0f;
#pragma unroll 8
    for (int d = 0; d < HD_; d++) acc += q[qh * HD_ + d] * k[kh * HD_ + d];
    sc[qh][kh] = acc * 0.125f;  // 1/sqrt(64)
  }
  __syncthreads();
  if (t < H_) {  // softmax over kh
    float mx = -1e30f;
    for (int j = 0; j < H_; j++) mx = fmaxf(mx, sc[t][j]);
    float sum = 0.0f;
    for (int j = 0; j < H_; j++) {
      float e = expf(sc[t][j] - mx);
      at[t][j] = e;
      sum += e;
    }
    float inv = 1.0f / sum;
    for (int j = 0; j < H_; j++) at[t][j] *= inv;
  }
  __syncthreads();
  for (int i = t; i < D_; i += 256) {
    int h = i >> 6, hd = i & 63;
    float acc = 0.0f;
#pragma unroll
    for (int kk = 0; kk < H_; kk++) acc += at[h][kk] * v[kk * HD_ + hd];
    int r = h * 64 + (s >> 4);
    int c = ((s & 15) << 6) + hd;
    A16[(size_t)b * S_ * D_ + (size_t)r * D_ + c] = (bf16)acc;
  }
}

// ---------------- gating: softmax(mean_s a @ Wg + bg) ----------------
__global__ __launch_bounds__(256)
void gate_kernel(const bf16* __restrict__ A16, const float* __restrict__ Wg,
                 const float* __restrict__ bg, float* __restrict__ gates) {
  __shared__ float meanv[D_];
  __shared__ float logit[E_];
  int b = blockIdx.x, t = threadIdx.x;
  const bf16* Ab = A16 + (size_t)b * S_ * D_;
  for (int d = t; d < D_; d += 256) {
    float s = 0.0f;
    for (int r = 0; r < S_; r++) s += (float)Ab[(size_t)r * D_ + d];
    meanv[d] = s * (1.0f / (float)S_);
  }
  __syncthreads();
  if (t < E_) {  // deterministic single-thread dot per expert (tiny)
    float s = bg[t];
    for (int d = 0; d < D_; d++) s += meanv[d] * Wg[d * E_ + t];
    logit[t] = s;
  }
  __syncthreads();
  if (t == 0) {
    float mx = -1e30f;
    for (int e = 0; e < E_; e++) mx = fmaxf(mx, logit[e]);
    float sum = 0.0f;
    float buf[E_];
    for (int e = 0; e < E_; e++) {
      buf[e] = expf(logit[e] - mx);
      sum += buf[e];
    }
    float inv = 1.0f / sum;
    for (int e = 0; e < E_; e++) gates[b * E_ + e] = buf[e] * inv;
  }
}

extern "C" void kernel_launch(void* const* d_in, const int* in_sizes, int n_in,
                              void* d_out, int out_size, void* d_ws, size_t ws_size,
                              hipStream_t stream) {
  const float* x  = (const float*)d_in[0];
  const float* Wq = (const float*)d_in[1];
  const float* bq = (const float*)d_in[2];
  const float* Wk = (const float*)d_in[3];
  const float* bk = (const float*)d_in[4];
  const float* Wv = (const float*)d_in[5];
  const float* bv = (const float*)d_in[6];
  const float* Wg = (const float*)d_in[7];
  const float* bg = (const float*)d_in[8];
  const float* W1 = (const float*)d_in[9];
  const float* b1 = (const float*)d_in[10];
  const float* W2 = (const float*)d_in[11];
  const float* b2 = (const float*)d_in[12];
  float* out = (float*)d_out;

  char* wp = (char*)d_ws;
  auto carve = [&](size_t bytes) -> char* {
    char* p = wp;
    wp += (bytes + 255) & ~(size_t)255;
    return p;
  };
  bf16*  X16    = (bf16*)carve((size_t)NT_ * D_ * sizeof(bf16));        // 8 MB
  bf16*  Wqkv16 = (bf16*)carve((size_t)D_ * 3 * D_ * sizeof(bf16));     // 6 MB
  float* bqkv   = (float*)carve((size_t)3 * D_ * sizeof(float));
  bf16*  W1_16  = (bf16*)carve((size_t)E_ * D_ * D_ * sizeof(bf16));    // 16 MB
  bf16*  W2_16  = (bf16*)carve((size_t)E_ * D_ * D_ * sizeof(bf16));    // 16 MB
  float* QKV    = (float*)carve((size_t)NT_ * 3 * D_ * sizeof(float));  // 48 MB
  bf16*  A16    = (bf16*)carve((size_t)NT_ * D_ * sizeof(bf16));        // 8 MB
  bf16*  H16    = (bf16*)carve((size_t)NT_ * D_ * sizeof(bf16));        // 8 MB
  float* gates  = (float*)carve((size_t)B_ * E_ * sizeof(float));

  // precision conversion / weight packing
  cvt_f32_bf16<<<1024, 256, 0, stream>>>(x, X16, NT_ * D_);
  pack_qkv<<<1024, 256, 0, stream>>>(Wq, Wk, Wv, bq, bk, bv, Wqkv16, bqkv);
  cvt_f32_bf16<<<2048, 256, 0, stream>>>(W1, W1_16, E_ * D_ * D_);
  cvt_f32_bf16<<<2048, 256, 0, stream>>>(W2, W2_16, E_ * D_ * D_);

  // fused QKV projection: [4096,1024] @ [1024,3072] + bias -> fp32
  dim3 g0(3 * D_ / 128, NT_ / 128);
  gemm_bf16<0><<<g0, 256, 0, stream>>>(X16, Wqkv16, bqkv, QKV, nullptr,
                                       nullptr, 0, 0, NT_, 3 * D_, D_);

  attn_kernel<<<NT_, 256, 0, stream>>>(QKV, A16);
  gate_kernel<<<B_, 256, 0, stream>>>(A16, Wg, bg, gates);

  // dense MoE: per expert, up-proj + gelu (bf16) then gate-scaled down-proj
  dim3 g1(D_ / 128, NT_ / 128);
  for (int e = 0; e < E_; e++) {
    gemm_bf16<1><<<g1, 256, 0, stream>>>(
        A16, W1_16 + (size_t)e * D_ * D_, b1 + (size_t)e * D_,
        nullptr, H16, nullptr, 0, 0, NT_, D_, D_);
    gemm_bf16<2><<<g1, 256, 0, stream>>>(
        H16, W2_16 + (size_t)e * D_ * D_, b2 + (size_t)e * D_,
        out, nullptr, gates, e, (e > 0) ? 1 : 0, NT_, D_, D_);
  }
}
